// DiTeBlock_46454366274176
// MI455X (gfx1250) — compile-verified, matrix operand order checked by reading, not run on hardware
//
#include <hip/hip_runtime.h>
#include <hip/hip_bf16.h>

// ---------------- problem constants (match reference) ----------------
constexpr int NN   = 768;    // nodes
constexpr int BB   = 16;     // graphs
constexpr int HH   = 256;    // node hidden
constexpr int EHH  = 128;    // edge hidden
constexpr int DST  = 128;    // dist features
constexpr int NHD  = 8;      // heads
constexpr int DH   = 32;     // head dim
constexpr int SW   = 1024;   // swiglu width
constexpr int EE   = 24576;  // edges

// ---------------- WMMA types ----------------
typedef __attribute__((ext_vector_type(16))) __bf16 v16bf;
typedef __attribute__((ext_vector_type(8)))  __bf16 v8bf;
typedef __attribute__((ext_vector_type(8)))  float  v8f;

__device__ __forceinline__ float siluf(float v) { return v / (1.f + __expf(-v)); }

// Load one 16x32 bf16 WMMA operand fragment (A layout; B uses the same lane
// pattern against the pre-transposed weight). ISA 16-bit A layout:
// lanes 0-15 hold K = {k0..k0+7, k0+16..k0+23}; lanes 16-31 hold the other halves.
__device__ __forceinline__ v16bf load_frag(const __bf16* base, int ld, int row,
                                           int k0, int half) {
  const __bf16* p = base + (size_t)row * ld + k0 + half * 8;
  v8bf lo = *reinterpret_cast<const v8bf*>(p);        // 16B contiguous
  v8bf hi = *reinterpret_cast<const v8bf*>(p + 16);   // 16B contiguous
  return __builtin_shufflevector(lo, hi, 0,1,2,3,4,5,6,7,8,9,10,11,12,13,14,15);
}

// C[M,N] = act(A[M,K] @ Bt[N,K]^T + bias). A, Bt bf16 row-major (lda/ldb elems).
// Exactly one of Cf/Cb non-null. act: 0 = none, 1 = SiLU.
// Each wave computes a 32 x (16*TN) macro-tile: 2 A fragments and TN B
// fragments are loaded up-front each k-step (12 b128 loads for 8 WMMAs at
// TN=4), then 2*TN WMMAs run on independent accumulators so the XDL pipe
// stays busy past the WMMA->WMMA RAW hazard while later loads are in flight.
template <int TN>
__global__ void k_gemm_bf16(const __hip_bfloat16* __restrict__ A, int lda,
                            const __hip_bfloat16* __restrict__ Bt, int ldb,
                            float* __restrict__ Cf, __hip_bfloat16* __restrict__ Cb,
                            int ldc, const float* __restrict__ bias, int act,
                            int M, int N, int K) {
  int wave = blockIdx.x * (blockDim.x >> 5) + (threadIdx.x >> 5);
  int lane = threadIdx.x & 31;
  int tn = N / (16 * TN);
  if (wave >= (M >> 5) * tn) return;       // wave-uniform exit (EXEC all-1 inside)
  int tmi = wave / tn, tni = wave % tn;
  int half = lane >> 4, l15 = lane & 15;
  const __bf16* Ab = reinterpret_cast<const __bf16*>(A);
  const __bf16* Bb = reinterpret_cast<const __bf16*>(Bt);
  int arow0 = tmi * 32 + l15;
  v8f acc[2][TN] = {};
  for (int k0 = 0; k0 < K; k0 += 32) {
    v16bf a0 = load_frag(Ab, lda, arow0, k0, half);
    v16bf a1 = load_frag(Ab, lda, arow0 + 16, k0, half);
    v16bf b[TN];
#pragma unroll
    for (int t = 0; t < TN; ++t)
      b[t] = load_frag(Bb, ldb, (tni * TN + t) * 16 + l15, k0, half);
#pragma unroll
    for (int t = 0; t < TN; ++t)
      acc[0][t] = __builtin_amdgcn_wmma_f32_16x16x32_bf16(false, a0, false, b[t],
                                                          (short)0, acc[0][t], false, false);
#pragma unroll
    for (int t = 0; t < TN; ++t)
      acc[1][t] = __builtin_amdgcn_wmma_f32_16x16x32_bf16(false, a1, false, b[t],
                                                          (short)0, acc[1][t], false, false);
  }
#pragma unroll
  for (int mt = 0; mt < 2; ++mt) {
#pragma unroll
    for (int t = 0; t < TN; ++t) {
      int colg = (tni * TN + t) * 16 + l15;
      float bv = bias ? bias[colg] : 0.f;
#pragma unroll
      for (int r = 0; r < 8; ++r) {
        int rowg = tmi * 32 + mt * 16 + r + half * 8;  // 16x16 f32 C/D VGPR layout
        float v = acc[mt][t][r] + bv;
        if (act == 1) v = siluf(v);
        size_t idx = (size_t)rowg * ldc + colg;
        if (Cf) Cf[idx] = v; else Cb[idx] = __float2bfloat16(v);
      }
    }
  }
}

// ---------------- elementwise / reduction kernels ----------------
__global__ void k_zero(float* p, long n) {
  long i = (long)blockIdx.x * blockDim.x + threadIdx.x;
  if (i < n) p[i] = 0.f;
}

// per-graph adaLN: out[B,OUT] = silu(t[B,IN]) @ W[IN,OUT] + b
__global__ void k_graph_linear(const float* t, const float* W, const float* b,
                               float* out, int Bn, int IN, int OUT) {
  int i = blockIdx.x * blockDim.x + threadIdx.x;
  if (i >= Bn * OUT) return;
  int g = i / OUT, o = i % OUT;
  float acc = b[o];
  for (int k = 0; k < IN; ++k) acc += siluf(t[g * IN + k]) * W[k * OUT + o];
  out[i] = acc;
}

// acc layout: [sum(B), sumsq(B), cnt(B), mean(B), rstd(B)]
__global__ void k_bln_stats(const float* x, int ld, const int* seg, float* acc,
                            int R, int C, int Bn) {
  int r = blockIdx.x * blockDim.x + threadIdx.x;
  if (r >= R) return;
  const float* row = x + (size_t)r * ld;
  float s = 0.f, ss = 0.f;
  for (int c = 0; c < C; ++c) { float v = row[c]; s += v; ss += v * v; }
  int g = seg[r];
  atomicAdd(&acc[g], s); atomicAdd(&acc[Bn + g], ss); atomicAdd(&acc[2 * Bn + g], (float)C);
}
__global__ void k_bln_final(float* a, int Bn, float eps) {
  int g = blockIdx.x * blockDim.x + threadIdx.x;
  if (g >= Bn) return;
  float cnt = fmaxf(a[2 * Bn + g], 1.f);
  float mean = a[g] / cnt;
  float var = a[Bn + g] / cnt - mean * mean;
  a[3 * Bn + g] = mean;
  a[4 * Bn + g] = rsqrtf(fmaxf(var, 0.f) + eps);
}
__global__ void k_bln_apply_mod_bf(const float* x, int ld, const int* seg,
                                   const float* acc, int Bn, const float* mod,
                                   int ms, int sc, int sh,
                                   __hip_bfloat16* out, int R, int C) {
  int i = blockIdx.x * blockDim.x + threadIdx.x;
  if (i >= R * C) return;
  int r = i / C, c = i % C, g = seg[r];
  float v = (x[(size_t)r * ld + c] - acc[3 * Bn + g]) * acc[4 * Bn + g];
  v = v * (1.f + mod[g * ms + sc + c]) + mod[g * ms + sh + c];
  out[i] = __float2bfloat16(v);
}
__global__ void k_bln_apply_mod_f32(const float* x, int ld, const int* seg,
                                    const float* acc, int Bn, const float* mod,
                                    int ms, int sc, int sh,
                                    float* out, int R, int C) {
  int i = blockIdx.x * blockDim.x + threadIdx.x;
  if (i >= R * C) return;
  int r = i / C, c = i % C, g = seg[r];
  float v = (x[(size_t)r * ld + c] - acc[3 * Bn + g]) * acc[4 * Bn + g];
  out[i] = v * (1.f + mod[g * ms + sc + c]) + mod[g * ms + sh + c];
}
__global__ void k_bln_affine_bf(const float* x, int ld, const int* seg,
                                const float* acc, int Bn, const float* gamma,
                                const float* beta, __hip_bfloat16* out, int R, int C) {
  int i = blockIdx.x * blockDim.x + threadIdx.x;
  if (i >= R * C) return;
  int r = i / C, c = i % C, g = seg[r];
  float v = (x[(size_t)r * ld + c] - acc[3 * Bn + g]) * acc[4 * Bn + g];
  if (gamma) v = v * gamma[c] + beta[c];
  out[i] = __float2bfloat16(v);
}

__global__ void k_build_m_in(const __hip_bfloat16* xn, const __hip_bfloat16* en,
                             const float* dist, const int* src, const int* tgt,
                             __hip_bfloat16* m_in, long total) {
  long i = (long)blockIdx.x * blockDim.x + threadIdx.x;
  if (i >= total) return;
  int e = (int)(i / (2 * HH + EHH + DST)), c = (int)(i % (2 * HH + EHH + DST));
  __hip_bfloat16 v;
  if (c < HH)                v = xn[(size_t)src[e] * HH + c];
  else if (c < 2 * HH)       v = xn[(size_t)tgt[e] * HH + (c - HH)];
  else if (c < 2 * HH + EHH) v = en[(size_t)e * EHH + (c - 2 * HH)];
  else                       v = __float2bfloat16(dist[(size_t)e * DST + (c - 2 * HH - EHH)]);
  m_in[i] = v;
}

__global__ void k_scatter_cnt(const int* src, float* cnt, int E) {
  int e = blockIdx.x * blockDim.x + threadIdx.x;
  if (e < E) atomicAdd(&cnt[src[e]], 1.f);
}
__global__ void k_scatter_add(const float* msg, const int* src, float* agg, long total) {
  long i = (long)blockIdx.x * blockDim.x + threadIdx.x;
  if (i >= total) return;
  int e = (int)(i / HH), c = (int)(i % HH);
  atomicAdd(&agg[(size_t)src[e] * HH + c], msg[i]);
}
__global__ void k_div_cnt_bf(const float* agg, const float* cnt,
                             __hip_bfloat16* out, long total) {
  long i = (long)blockIdx.x * blockDim.x + threadIdx.x;
  if (i >= total) return;
  out[i] = __float2bfloat16(agg[i] / fmaxf(cnt[i / HH], 1.f));
}

// bias[n,m] = (n==m) ? 0 : dot(silu(Z[n,m,:]), w_pair).
// One wave per (n,m); each lane owns 8 contiguous channels -> two b128 loads
// per lane on the 604 MB Z stream (the HBM-bandwidth floor of this op).
__global__ void k_bias_z(const float* __restrict__ Z, const float* __restrict__ w_pair,
                         float* __restrict__ bias) {
  int wid = (blockIdx.x * blockDim.x + threadIdx.x) >> 5;
  int lane = threadIdx.x & 31;
  if (wid >= NN * NN) return;
  const float4* z4 = reinterpret_cast<const float4*>(Z + (size_t)wid * HH) + lane * 2;
  const float4* w4 = reinterpret_cast<const float4*>(w_pair) + lane * 2;
  float4 za = z4[0], zb = z4[1];
  float4 wa = w4[0], wb = w4[1];
  float acc = siluf(za.x) * wa.x + siluf(za.y) * wa.y +
              siluf(za.z) * wa.z + siluf(za.w) * wa.w +
              siluf(zb.x) * wb.x + siluf(zb.y) * wb.y +
              siluf(zb.z) * wb.z + siluf(zb.w) * wb.w;
  for (int off = 16; off; off >>= 1) acc += __shfl_xor(acc, off, 32);
  if (lane == 0) { int n = wid / NN, m = wid % NN; bias[wid] = (n == m) ? 0.f : acc; }
}

// fused scale + mask + pair bias + softmax row -> bf16 attention
__global__ void k_softmax(const float* scores, const float* biasZ, const int* batch,
                          __hip_bfloat16* attn, float scale) {
  int row = blockIdx.x;                 // h*NN + i
  int i = row % NN;
  const float* s = scores + (size_t)row * NN;
  const float* bz = biasZ + (size_t)i * NN;
  int bi = batch[i];
  __shared__ float buf[NN];
  __shared__ float red[256];
  int t = threadIdx.x;
  float mx = -3.4e38f;
  for (int m = t; m < NN; m += 256) {
    float v = s[m] * scale + bz[m] + (batch[m] == bi ? 0.f : -1e30f);
    buf[m] = v; mx = fmaxf(mx, v);
  }
  red[t] = mx; __syncthreads();
  for (int o = 128; o; o >>= 1) { if (t < o) red[t] = fmaxf(red[t], red[t + o]); __syncthreads(); }
  mx = red[0]; __syncthreads();
  float sm = 0.f;
  for (int m = t; m < NN; m += 256) { float e = __expf(buf[m] - mx); buf[m] = e; sm += e; }
  red[t] = sm; __syncthreads();
  for (int o = 128; o; o >>= 1) { if (t < o) red[t] += red[t + o]; __syncthreads(); }
  float inv = 1.f / red[0];
  __hip_bfloat16* a = attn + (size_t)row * NN;
  for (int m = t; m < NN; m += 256) a[m] = __float2bfloat16(buf[m] * inv);
}

__global__ void k_f32_to_bf(const float* in, __hip_bfloat16* out, long n) {
  long i = (long)blockIdx.x * blockDim.x + threadIdx.x;
  if (i < n) out[i] = __float2bfloat16(in[i]);
}
// out[c*R + r] = bf(in[r*ld + off + c])
__global__ void k_transpose_bf(const float* in, int ld, int off,
                               __hip_bfloat16* out, int R, int C) {
  int i = blockIdx.x * blockDim.x + threadIdx.x;
  if (i >= R * C) return;
  int r = i / C, c = i % C;
  out[(size_t)c * R + r] = __float2bfloat16(in[(size_t)r * ld + off + c]);
}
// weight transpose+convert: Wt[o*K+k] = bf(W[k*OUT+o])
__global__ void k_wT_bf(const float* W, __hip_bfloat16* Wt, int K, int OUT) {
  int i = blockIdx.x * blockDim.x + threadIdx.x;
  if (i >= K * OUT) return;
  int k = i / OUT, o = i % OUT;
  Wt[(size_t)o * K + k] = __float2bfloat16(W[i]);
}
// out = base + mod[seg[r]*ms+off+c] * val
__global__ void k_add_scale_mod(const float* base, const float* val, const float* mod,
                                const int* seg, int ms, int off,
                                float* out, int R, int C) {
  long i = (long)blockIdx.x * blockDim.x + threadIdx.x;
  if (i >= (long)R * C) return;
  int r = (int)(i / C), c = (int)(i % C);
  out[i] = base[i] + mod[seg[r] * ms + off + c] * val[i];
}
__global__ void k_ysum_bf(const float* y, const int* src, const int* tgt,
                          __hip_bfloat16* out, long total) {
  long i = (long)blockIdx.x * blockDim.x + threadIdx.x;
  if (i >= total) return;
  int e = (int)(i / HH), c = (int)(i % HH);
  out[i] = __float2bfloat16(y[(size_t)src[e] * HH + c] + y[(size_t)tgt[e] * HH + c]);
}
// out[r*S+c] = bf(silu(in[r*2S+c]) * in[r*2S+S+c])
__global__ void k_swiglu_bf(const float* in, __hip_bfloat16* out, int R, int S) {
  long i = (long)blockIdx.x * blockDim.x + threadIdx.x;
  if (i >= (long)R * S) return;
  int r = (int)(i / S), c = (int)(i % S);
  out[i] = __float2bfloat16(siluf(in[(size_t)r * 2 * S + c]) * in[(size_t)r * 2 * S + S + c]);
}
__global__ void k_concat2_bf(const float* a, const float* b, __hip_bfloat16* out,
                             int R, int C1, int C2) {
  long i = (long)blockIdx.x * blockDim.x + threadIdx.x;
  int C = C1 + C2;
  if (i >= (long)R * C) return;
  int r = (int)(i / C), c = (int)(i % C);
  out[i] = __float2bfloat16(c < C1 ? a[(size_t)r * C1 + c] : b[(size_t)r * C2 + (c - C1)]);
}

// ---------------- host orchestration ----------------
static inline int cdiv(long a, long b) { return (int)((a + b - 1) / b); }

static void gemm(hipStream_t s, const __hip_bfloat16* A, int lda,
                 const __hip_bfloat16* Bt, int ldb, float* Cf, __hip_bfloat16* Cb,
                 int ldc, const float* bias, int act, int M, int N, int K) {
  if (N % 64 == 0) {
    int tiles = (M / 32) * (N / 64);
    k_gemm_bf16<4><<<cdiv(tiles, 8), 256, 0, s>>>(A, lda, Bt, ldb, Cf, Cb, ldc, bias, act, M, N, K);
  } else {
    int tiles = (M / 32) * (N / 32);
    k_gemm_bf16<2><<<cdiv(tiles, 8), 256, 0, s>>>(A, lda, Bt, ldb, Cf, Cb, ldc, bias, act, M, N, K);
  }
}

extern "C" void kernel_launch(void* const* d_in, const int* in_sizes, int n_in,
                              void* d_out, int out_size, void* d_ws, size_t ws_size,
                              hipStream_t stream) {
  // inputs (setup_inputs order; params flattened in dict insertion order)
  const int*   batch      = (const int*)  d_in[0];
  const float* x          = (const float*)d_in[1];
  const float* t_emb_h    = (const float*)d_in[2];
  const float* edge_attr  = (const float*)d_in[3];
  const int*   edge_index = (const int*)  d_in[4];
  const float* t_emb_e    = (const float*)d_in[5];
  const float* dist       = (const float*)d_in[6];
  const int*   edge_batch = (const int*)  d_in[7];
  const float* Z          = (const float*)d_in[8];
  const float* W_ada  = (const float*)d_in[9],  *b_ada  = (const float*)d_in[10];
  const float* W_adae = (const float*)d_in[11], *b_adae = (const float*)d_in[12];
  const float* W_fe1  = (const float*)d_in[13], *b_fe1  = (const float*)d_in[14];
  const float* W_fe2  = (const float*)d_in[15], *b_fe2  = (const float*)d_in[16];
  const float* W_qkv  = (const float*)d_in[17];
  const float* W_out  = (const float*)d_in[18];
  const float* w_pair = (const float*)d_in[19];
  const float* W_e0   = (const float*)d_in[20];
  const float* W_e1   = (const float*)d_in[21];
  const float* g_ffn  = (const float*)d_in[22], *b_ffn  = (const float*)d_in[23];
  const float* W_ffn1 = (const float*)d_in[24], *W_ffn2 = (const float*)d_in[25];
  const float* g_ffne = (const float*)d_in[26], *b_ffne = (const float*)d_in[27];
  const float* W_ffne1= (const float*)d_in[28], *W_ffne2= (const float*)d_in[29];
  const int* src = edge_index;
  const int* tgt = edge_index + EE;
  float* out_x = (float*)d_out;
  float* out_e = (float*)d_out + (size_t)NN * HH;

  // bump allocator over d_ws (~270 MB used)
  char* wp = (char*)d_ws;
  auto alloc = [&](size_t bytes) -> char* {
    char* p = wp; wp += (bytes + 255) & ~(size_t)255; return p;
  };
  typedef __hip_bfloat16 bf;
  float* mod      = (float*)alloc(sizeof(float) * BB * 6 * HH);
  float* emod     = (float*)alloc(sizeof(float) * BB * 6 * EHH);
  float* blnmem   = (float*)alloc(sizeof(float) * 8 * 5 * BB);   // 8 BLN instances
  // weight transposes (bf16)
  bf* W_fe1t  = (bf*)alloc(2ull * 256 * 768);
  bf* W_fe2t  = (bf*)alloc(2ull * 256 * 256);
  bf* W_qkvt  = (bf*)alloc(2ull * 768 * 256);
  bf* W_outt  = (bf*)alloc(2ull * 256 * 256);
  bf* W_e0t   = (bf*)alloc(2ull * 128 * 256);
  bf* W_e1t   = (bf*)alloc(2ull * 128 * 256);
  bf* W_ffn1t = (bf*)alloc(2ull * 2048 * 256);
  bf* W_ffn2t = (bf*)alloc(2ull * 256 * 1024);
  bf* W_ffne1t= (bf*)alloc(2ull * 256 * 128);
  bf* W_ffne2t= (bf*)alloc(2ull * 128 * 128);
  // activations
  bf*    x_norm  = (bf*)   alloc(2ull * NN * HH);
  bf*    e_norm  = (bf*)   alloc(2ull * EE * EHH);
  bf*    m_in    = (bf*)   alloc(2ull * EE * (2 * HH + EHH + DST));
  bf*    h1_bf   = (bf*)   alloc(2ull * EE * HH);
  float* messages= (float*)alloc(4ull * EE * HH);
  float* cntN    = (float*)alloc(4ull * NN);
  float* x_agg   = (float*)alloc(4ull * NN * HH);
  bf*    x_agg_bf= (bf*)   alloc(2ull * NN * HH);
  float* qkv     = (float*)alloc(4ull * NN * 3 * HH);
  bf*    q_bf    = (bf*)   alloc(2ull * NN * HH);
  bf*    k_bf    = (bf*)   alloc(2ull * NN * HH);
  bf*    vt_bf   = (bf*)   alloc(2ull * HH * NN);
  float* scores  = (float*)alloc(4ull * NHD * NN * NN);
  float* biasZ   = (float*)alloc(4ull * NN * NN);
  bf*    attn_bf = (bf*)   alloc(2ull * NHD * NN * NN);
  float* outm    = (float*)alloc(4ull * NN * HH);
  bf*    out_bf  = (bf*)   alloc(2ull * NN * HH);
  float* yv      = (float*)alloc(4ull * NN * HH);
  float* x_new   = (float*)alloc(4ull * NN * HH);
  bf*    ysum_bf = (bf*)   alloc(2ull * EE * HH);
  float* e0out   = (float*)alloc(4ull * EE * EHH);
  float* edge1   = (float*)alloc(4ull * EE * EHH);
  float* xn      = (float*)alloc(4ull * NN * HH);
  bf*    xn2_bf  = (bf*)   alloc(2ull * NN * HH);
  float* ffn1    = (float*)alloc(4ull * NN * 2 * SW);
  bf*    hsw_bf  = (bf*)   alloc(2ull * NN * SW);
  float* ffn2    = (float*)alloc(4ull * NN * HH);
  bf*    ecat_bf = (bf*)   alloc(2ull * EE * (EHH + DST));
  float* e_in    = (float*)alloc(4ull * EE * EHH);
  float* en      = (float*)alloc(4ull * EE * EHH);
  bf*    en2_bf  = (bf*)   alloc(2ull * EE * EHH);
  float* effn1   = (float*)alloc(4ull * EE * 2 * EHH);
  bf*    ehsw_bf = (bf*)   alloc(2ull * EE * EHH);
  float* effn2   = (float*)alloc(4ull * EE * EHH);

  float* bln[8];
  for (int i = 0; i < 8; ++i) bln[i] = blnmem + i * 5 * BB;
  const int T = 256;
  #define GRID(n) cdiv((long)(n), T), T, 0, stream

  // 0) weights -> bf16 transposed
  k_wT_bf<<<GRID(768 * 256)>>>(W_fe1,  W_fe1t,  768, 256);
  k_wT_bf<<<GRID(256 * 256)>>>(W_fe2,  W_fe2t,  256, 256);
  k_wT_bf<<<GRID(256 * 768)>>>(W_qkv,  W_qkvt,  256, 768);
  k_wT_bf<<<GRID(256 * 256)>>>(W_out,  W_outt,  256, 256);
  k_wT_bf<<<GRID(256 * 128)>>>(W_e0,   W_e0t,   256, 128);
  k_wT_bf<<<GRID(256 * 128)>>>(W_e1,   W_e1t,   256, 128);
  k_wT_bf<<<GRID(256 * 2048)>>>(W_ffn1, W_ffn1t, 256, 2048);
  k_wT_bf<<<GRID(1024 * 256)>>>(W_ffn2, W_ffn2t, 1024, 256);
  k_wT_bf<<<GRID(128 * 256)>>>(W_ffne1, W_ffne1t, 128, 256);
  k_wT_bf<<<GRID(128 * 128)>>>(W_ffne2, W_ffne2t, 128, 128);

  // 1) per-graph adaLN modulation
  k_graph_linear<<<GRID(BB * 6 * HH)>>>(t_emb_h, W_ada, b_ada, mod, BB, HH, 6 * HH);
  k_graph_linear<<<GRID(BB * 6 * EHH)>>>(t_emb_e, W_adae, b_adae, emod, BB, EHH, 6 * EHH);
  k_zero<<<GRID(8 * 5 * BB)>>>(blnmem, 8 * 5 * BB);

  // 2) node / edge BLN + modulation
  k_bln_stats<<<GRID(NN)>>>(x, HH, batch, bln[0], NN, HH, BB);
  k_bln_final<<<GRID(BB)>>>(bln[0], BB, 1e-6f);
  k_bln_apply_mod_bf<<<GRID(NN * HH)>>>(x, HH, batch, bln[0], BB, mod, 6 * HH, HH, 0, x_norm, NN, HH);
  k_bln_stats<<<GRID(EE)>>>(edge_attr, EHH, edge_batch, bln[1], EE, EHH, BB);
  k_bln_final<<<GRID(BB)>>>(bln[1], BB, 1e-6f);
  k_bln_apply_mod_bf<<<GRID(EE * EHH)>>>(edge_attr, EHH, edge_batch, bln[1], BB, emod, 6 * EHH, EHH, 0, e_norm, EE, EHH);

  // 3) edge message MLP (the big GEMMs)
  k_build_m_in<<<GRID((long)EE * (2 * HH + EHH + DST))>>>(x_norm, e_norm, dist, src, tgt, m_in, (long)EE * (2 * HH + EHH + DST));
  gemm(stream, m_in, 768, W_fe1t, 768, nullptr, h1_bf, HH, b_fe1, 1, EE, HH, 768);
  gemm(stream, h1_bf, HH, W_fe2t, HH, messages, nullptr, HH, b_fe2, 0, EE, HH, HH);

  // 4) scatter-mean
  k_zero<<<GRID(NN)>>>(cntN, NN);
  k_zero<<<GRID(NN * HH)>>>(x_agg, (long)NN * HH);
  k_scatter_cnt<<<GRID(EE)>>>(src, cntN, EE);
  k_scatter_add<<<GRID((long)EE * HH)>>>(messages, src, x_agg, (long)EE * HH);
  k_div_cnt_bf<<<GRID(NN * HH)>>>(x_agg, cntN, x_agg_bf, (long)NN * HH);

  // 5) QKV + qk graph-norm
  gemm(stream, x_agg_bf, HH, W_qkvt, HH, qkv, nullptr, 3 * HH, nullptr, 0, NN, 3 * HH, HH);
  k_bln_stats<<<GRID(NN)>>>(qkv, 3 * HH, batch, bln[2], NN, HH, BB);
  k_bln_final<<<GRID(BB)>>>(bln[2], BB, 1e-6f);
  k_bln_affine_bf<<<GRID(NN * HH)>>>(qkv, 3 * HH, batch, bln[2], BB, nullptr, nullptr, q_bf, NN, HH);
  k_bln_stats<<<GRID(NN)>>>(qkv + HH, 3 * HH, batch, bln[3], NN, HH, BB);
  k_bln_final<<<GRID(BB)>>>(bln[3], BB, 1e-6f);
  k_bln_affine_bf<<<GRID(NN * HH)>>>(qkv + HH, 3 * HH, batch, bln[3], BB, nullptr, nullptr, k_bf, NN, HH);
  k_transpose_bf<<<GRID(NN * HH)>>>(qkv, 3 * HH, 2 * HH, vt_bf, NN, HH);   // V^T

  // 6) attention: pair bias (604 MB stream), QK^T per head, softmax, attn@V
  k_bias_z<<<cdiv((long)NN * NN * 32, T), T, 0, stream>>>(Z, w_pair, biasZ);
  for (int h = 0; h < NHD; ++h)
    gemm(stream, q_bf + h * DH, HH, k_bf + h * DH, HH,
         scores + (size_t)h * NN * NN, nullptr, NN, nullptr, 0, NN, NN, DH);
  k_softmax<<<NHD * NN, T, 0, stream>>>(scores, biasZ, batch, attn_bf, 0.1767766953f);
  for (int h = 0; h < NHD; ++h)
    gemm(stream, attn_bf + (size_t)h * NN * NN, NN, vt_bf + (size_t)h * DH * NN, NN,
         outm + h * DH, nullptr, HH, nullptr, 0, NN, DH, NN);

  // 7) output proj + residuals
  k_f32_to_bf<<<GRID(NN * HH)>>>(outm, out_bf, (long)NN * HH);
  gemm(stream, out_bf, HH, W_outt, HH, yv, nullptr, HH, nullptr, 0, NN, HH, HH);
  k_add_scale_mod<<<GRID(NN * HH)>>>(x, yv, mod, batch, 6 * HH, 2 * HH, x_new, NN, HH);
  k_ysum_bf<<<GRID((long)EE * HH)>>>(yv, src, tgt, ysum_bf, (long)EE * HH);
  gemm(stream, ysum_bf, HH, W_e0t, HH, e0out, nullptr, EHH, nullptr, 0, EE, EHH, HH);
  k_add_scale_mod<<<GRID((long)EE * EHH)>>>(edge_attr, e0out, emod, edge_batch, 6 * EHH, 2 * EHH, edge1, EE, EHH);

  // 8) node SwiGLU FFN
  k_bln_stats<<<GRID(NN)>>>(x_new, HH, batch, bln[4], NN, HH, BB);
  k_bln_final<<<GRID(BB)>>>(bln[4], BB, 1e-6f);
  k_bln_apply_mod_f32<<<GRID(NN * HH)>>>(x_new, HH, batch, bln[4], BB, mod, 6 * HH, 4 * HH, 3 * HH, xn, NN, HH);
  k_bln_stats<<<GRID(NN)>>>(xn, HH, batch, bln[5], NN, HH, BB);
  k_bln_final<<<GRID(BB)>>>(bln[5], BB, 1e-5f);
  k_bln_affine_bf<<<GRID(NN * HH)>>>(xn, HH, batch, bln[5], BB, g_ffn, b_ffn, xn2_bf, NN, HH);
  gemm(stream, xn2_bf, HH, W_ffn1t, HH, ffn1, nullptr, 2 * SW, nullptr, 0, NN, 2 * SW, HH);
  k_swiglu_bf<<<GRID((long)NN * SW)>>>(ffn1, hsw_bf, NN, SW);
  gemm(stream, hsw_bf, SW, W_ffn2t, SW, ffn2, nullptr, HH, nullptr, 0, NN, HH, SW);
  k_add_scale_mod<<<GRID(NN * HH)>>>(x_new, ffn2, mod, batch, 6 * HH, 5 * HH, out_x, NN, HH);

  // 9) edge SwiGLU FFN
  k_concat2_bf<<<GRID((long)EE * (EHH + DST))>>>(edge1, dist, ecat_bf, EE, EHH, DST);
  gemm(stream, ecat_bf, EHH + DST, W_e1t, EHH + DST, e_in, nullptr, EHH, nullptr, 0, EE, EHH, EHH + DST);
  k_bln_stats<<<GRID(EE)>>>(e_in, EHH, edge_batch, bln[6], EE, EHH, BB);
  k_bln_final<<<GRID(BB)>>>(bln[6], BB, 1e-6f);
  k_bln_apply_mod_f32<<<GRID((long)EE * EHH)>>>(e_in, EHH, edge_batch, bln[6], BB, emod, 6 * EHH, 4 * EHH, 3 * EHH, en, EE, EHH);
  k_bln_stats<<<GRID(EE)>>>(en, EHH, edge_batch, bln[7], EE, EHH, BB);
  k_bln_final<<<GRID(BB)>>>(bln[7], BB, 1e-5f);
  k_bln_affine_bf<<<GRID((long)EE * EHH)>>>(en, EHH, edge_batch, bln[7], BB, g_ffne, b_ffne, en2_bf, EE, EHH);
  gemm(stream, en2_bf, EHH, W_ffne1t, EHH, effn1, nullptr, 2 * EHH, nullptr, 0, EE, 2 * EHH, EHH);
  k_swiglu_bf<<<GRID((long)EE * EHH)>>>(effn1, ehsw_bf, EE, EHH);
  gemm(stream, ehsw_bf, EHH, W_ffne2t, EHH, effn2, nullptr, EHH, nullptr, 0, EE, EHH, EHH);
  k_add_scale_mod<<<GRID((long)EE * EHH)>>>(edge1, effn2, emod, edge_batch, 6 * EHH, 5 * EHH, out_e, EE, EHH);
  #undef GRID
}
